// MaskedAttentionLayer_82987358094104
// MI455X (gfx1250) — matrix operand order repaired and checked by reading, not compiled, not run on hardware
//
#include <hip/hip_runtime.h>

// ---------------------------------------------------------------------------
// Types for CDNA5 WMMA (wave32): v_wmma_f32_16x16x32_bf16
// ---------------------------------------------------------------------------
typedef __attribute__((ext_vector_type(16))) __bf16 v16bf;
typedef __attribute__((ext_vector_type(8)))  __bf16 v8bf;
typedef __attribute__((ext_vector_type(8)))  float  v8f;

__device__ __forceinline__ __bf16 f32_to_bf16(float f) {
  unsigned int u = __builtin_bit_cast(unsigned int, f);
  u += 0x7FFFu + ((u >> 16) & 1u);
  unsigned short h = (unsigned short)(u >> 16);
  return __builtin_bit_cast(__bf16, h);
}

// Low 32 bits of a generic pointer into LDS == LDS byte offset (ISA 10.2:
// flat->LDS mapping truncates to addr[31:0]).
__device__ __forceinline__ unsigned lds_addr32(const void* p) {
  return (unsigned)(unsigned long long)p;
}

// Async DMA: global (16B per lane) -> LDS, tracked by ASYNCcnt.
__device__ __forceinline__ void async_load_b128(unsigned lds, const void* gaddr) {
  asm volatile("global_load_async_to_lds_b128 %0, %1, off"
               :: "v"(lds), "v"((unsigned long long)gaddr)
               : "memory");
}

// ---------------------------------------------------------------------------
// f32 -> bf16 convert
// ---------------------------------------------------------------------------
__global__ __launch_bounds__(256) void cvt_f32_bf16(const float* __restrict__ in,
                                                    __bf16* __restrict__ out, int n) {
  int i = blockIdx.x * blockDim.x + threadIdx.x;
  int stride = gridDim.x * blockDim.x;
  for (; i < n; i += stride) out[i] = f32_to_bf16(in[i]);
}

// ---------------------------------------------------------------------------
// Tiled WMMA GEMM:  C[M,N] = A[M,K] * B[N,K]^T  (+ epilogue)
//   MODE  0: C += bias[n]                       (linear layer)
//   MODE  1: C = (C - (n<=m ? 1e11 : 0)) / 64   (masked scores)
//   OUTBF 1: write bf16, OUTBF 0: write f32
// 128x128 block tile, BK=32, 8 waves (wave tile 64x32 = 4x2 WMMA tiles).
// Global->LDS staging uses async DMA with double-buffered LDS so tile k+1
// streams in while tile k feeds the matrix cores.
// ---------------------------------------------------------------------------
#define BM 128
#define BN 128
#define BK 32
#define LDSS 40  // LDS row stride in bf16 elements (32 + 8 pad)

template <int MODE, int OUTBF>
__global__ __launch_bounds__(256) void gemm_bt(const __bf16* __restrict__ A,
                                               const __bf16* __restrict__ B,
                                               const float* __restrict__ bias,
                                               float* __restrict__ outF,
                                               __bf16* __restrict__ outB,
                                               int M, int N, int K) {
  __shared__ __bf16 sA[2][BM * LDSS];
  __shared__ __bf16 sB[2][BN * LDSS];

  const int tid    = threadIdx.x;
  const int lane   = tid & 31;
  const int wave   = tid >> 5;
  const int waveM  = wave & 1;   // 2 x 64 rows
  const int waveN  = wave >> 1;  // 4 x 32 cols
  const int l16    = lane & 15;
  const int halfK  = lane >> 4;
  const int blockM = blockIdx.y * BM;
  const int blockN = blockIdx.x * BN;

  // loader geometry: each thread DMAs one 16B chunk per matrix per round
  const int ldRow = tid >> 2;        // 0..63
  const int ldCol = (tid & 3) * 8;   // 0,8,16,24

  // issue one tile's async loads (4 per thread => 4 ASYNCcnt ticks per wave)
  auto issue_tile = [&](int k0, int buf) {
#pragma unroll
    for (int r = 0; r < 2; ++r) {
      const int row = r * 64 + ldRow;
      async_load_b128(lds_addr32(&sA[buf][row * LDSS + ldCol]),
                      &A[(size_t)(blockM + row) * K + k0 + ldCol]);
      async_load_b128(lds_addr32(&sB[buf][row * LDSS + ldCol]),
                      &B[(size_t)(blockN + row) * K + k0 + ldCol]);
    }
  };

  v8f acc[4][2];
#pragma unroll
  for (int mi = 0; mi < 4; ++mi)
#pragma unroll
    for (int ni = 0; ni < 2; ++ni)
#pragma unroll
      for (int e = 0; e < 8; ++e) acc[mi][ni][e] = 0.0f;

  const int nk = K / BK;
  issue_tile(0, 0);

  for (int kt = 0; kt < nk; ++kt) {
    const int buf = kt & 1;
    if (kt + 1 < nk) {
      issue_tile((kt + 1) * BK, buf ^ 1);
      // newest 4 async loads (next tile) may stay in flight; this tile done
      asm volatile("s_wait_asynccnt 0x4" ::: "memory");
    } else {
      asm volatile("s_wait_asynccnt 0x0" ::: "memory");
    }
    __syncthreads();

    // A fragments (16x32 bf16): lanes 0-15 rows M=0..15 K 0..7/16..23,
    // lanes 16-31 same rows K 8..15/24..31.
    v16bf afrag[4];
#pragma unroll
    for (int mi = 0; mi < 4; ++mi) {
      const int row = waveM * 64 + mi * 16 + l16;
      v8bf lo = *(const v8bf*)&sA[buf][row * LDSS + halfK * 8];
      v8bf hi = *(const v8bf*)&sA[buf][row * LDSS + 16 + halfK * 8];
#pragma unroll
      for (int e = 0; e < 8; ++e) { afrag[mi][e] = lo[e]; afrag[mi][8 + e] = hi[e]; }
    }
    // B fragments (32x16 bf16): lane n holds column n; lanes 0-15 K 0..7/16..23,
    // lanes 16-31 K 8..15/24..31 (same order as A).
    v16bf bfrag[2];
#pragma unroll
    for (int ni = 0; ni < 2; ++ni) {
      const int row = waveN * 32 + ni * 16 + l16;
      v8bf lo = *(const v8bf*)&sB[buf][row * LDSS + halfK * 8];
      v8bf hi = *(const v8bf*)&sB[buf][row * LDSS + 16 + halfK * 8];
#pragma unroll
      for (int e = 0; e < 8; ++e) { bfrag[ni][e] = lo[e]; bfrag[ni][8 + e] = hi[e]; }
    }

#pragma unroll
    for (int mi = 0; mi < 4; ++mi)
#pragma unroll
      for (int ni = 0; ni < 2; ++ni)
        acc[mi][ni] = __builtin_amdgcn_wmma_f32_16x16x32_bf16(
            false, afrag[mi], false, bfrag[ni],
            (short)0, acc[mi][ni], false, false);

    __syncthreads();  // all waves done reading buf before it is re-filled
  }

  // Epilogue. C layout: VGPR r -> M = r + 8*(lane>=16), N = lane&15.
#pragma unroll
  for (int mi = 0; mi < 4; ++mi)
#pragma unroll
    for (int ni = 0; ni < 2; ++ni) {
      const int gn = blockN + waveN * 32 + ni * 16 + l16;
#pragma unroll
      for (int r = 0; r < 8; ++r) {
        const int gm = blockM + waveM * 64 + mi * 16 + halfK * 8 + r;
        float v = acc[mi][ni][r];
        if (MODE == 1) {
          v = (v - ((gn <= gm) ? 1e11f : 0.0f)) * 0.015625f;  // /sqrt(4096)
        } else {
          v += bias[gn];
        }
        const size_t idx = (size_t)gm * (size_t)N + (size_t)gn;
        if (OUTBF) outB[idx] = f32_to_bf16(v);
        else       outF[idx] = v;
      }
    }
}

// ---------------------------------------------------------------------------
// Softmax over axis 0 (per column), two-phase for parallelism.
// Phase 1: per (column, row-chunk) online max/sum partials.
// Phase 2: merge partials per column, write normalized bf16 P for own chunk.
// ---------------------------------------------------------------------------
__global__ __launch_bounds__(256) void softmax_part(const float* __restrict__ S,
                                                    float* __restrict__ Mp,
                                                    float* __restrict__ Dp,
                                                    int N, int CH) {
  const int j  = blockIdx.x * blockDim.x + threadIdx.x;
  const int c  = blockIdx.y;
  const int r0 = c * CH;
  float m = -3.4e38f, d = 0.0f;
  for (int i = r0; i < r0 + CH; ++i) {
    const float s = S[(size_t)i * N + j];
    const float mn = fmaxf(m, s);
    d = d * __expf(m - mn) + __expf(s - mn);
    m = mn;
  }
  Mp[(size_t)c * N + j] = m;
  Dp[(size_t)c * N + j] = d;
}

__global__ __launch_bounds__(256) void softmax_write(const float* __restrict__ S,
                                                     const float* __restrict__ Mp,
                                                     const float* __restrict__ Dp,
                                                     __bf16* __restrict__ P,
                                                     int N, int CH, int NC) {
  const int j = blockIdx.x * blockDim.x + threadIdx.x;
  const int c = blockIdx.y;
  float m = -3.4e38f, d = 0.0f;
  for (int cc = 0; cc < NC; ++cc) {
    const float mc = Mp[(size_t)cc * N + j];
    const float dc = Dp[(size_t)cc * N + j];
    const float mn = fmaxf(m, mc);
    d = d * __expf(m - mn) + dc * __expf(mc - mn);
    m = mn;
  }
  const float inv = 1.0f / d;
  const int r0 = c * CH;
  for (int i = r0; i < r0 + CH; ++i) {
    const float s = S[(size_t)i * N + j];
    P[(size_t)i * N + j] = f32_to_bf16(__expf(s - m) * inv);
  }
}

// ---------------------------------------------------------------------------
// Host-side orchestration
// ---------------------------------------------------------------------------
extern "C" void kernel_launch(void* const* d_in, const int* in_sizes, int n_in,
                              void* d_out, int out_size, void* d_ws, size_t ws_size,
                              hipStream_t stream) {
  (void)in_sizes; (void)n_in; (void)out_size; (void)ws_size;
  const float* x  = (const float*)d_in[0];
  const float* Wq = (const float*)d_in[1];
  const float* bq = (const float*)d_in[2];
  const float* Wk = (const float*)d_in[3];
  const float* bk = (const float*)d_in[4];
  const float* Wv = (const float*)d_in[5];
  const float* bv = (const float*)d_in[6];
  // d_in[7] (mask) unused: triu(k=1) mask applied analytically.

  const int IN_F = 2048, DK = 4096, OUT_F = 2048;

  char* ws = (char*)d_ws;
  size_t off = 0;
  auto alloc = [&](size_t bytes) -> char* {
    char* p = ws + off;
    off += (bytes + 255) & ~(size_t)255;
    return p;
  };
  __bf16* xb  = (__bf16*)alloc((size_t)DK * IN_F * 2);
  __bf16* Wqb = (__bf16*)alloc((size_t)DK * IN_F * 2);
  __bf16* Wkb = (__bf16*)alloc((size_t)DK * IN_F * 2);
  __bf16* Wvb = (__bf16*)alloc((size_t)OUT_F * DK * 2);
  __bf16* Qb  = (__bf16*)alloc((size_t)DK * DK * 2);
  __bf16* Kb  = (__bf16*)alloc((size_t)DK * DK * 2);
  float*  Sf  = (float*)alloc((size_t)DK * DK * 4);
  __bf16* Pb  = (__bf16*)alloc((size_t)DK * DK * 2);
  const int NC = 16, CH = DK / NC;
  float* Mp = (float*)alloc((size_t)NC * DK * 4);
  float* Dp = (float*)alloc((size_t)NC * DK * 4);

  // 1) convert inputs to bf16
  cvt_f32_bf16<<<512, 256, 0, stream>>>(x,  xb,  DK * IN_F);
  cvt_f32_bf16<<<512, 256, 0, stream>>>(Wq, Wqb, DK * IN_F);
  cvt_f32_bf16<<<512, 256, 0, stream>>>(Wk, Wkb, DK * IN_F);
  cvt_f32_bf16<<<512, 256, 0, stream>>>(Wv, Wvb, OUT_F * DK);

  const dim3 blk(256);
  // 2) Q = x @ Wq^T + bq ; K = x @ Wk^T + bk   (bf16 out)
  const dim3 gQK(DK / BN, DK / BM);
  gemm_bt<0, 1><<<gQK, blk, 0, stream>>>(xb, Wqb, bq, nullptr, Qb, DK, DK, IN_F);
  gemm_bt<0, 1><<<gQK, blk, 0, stream>>>(xb, Wkb, bk, nullptr, Kb, DK, DK, IN_F);

  // 3) S = (Q @ K^T - mask_penalty) / 64       (f32 out)
  gemm_bt<1, 0><<<gQK, blk, 0, stream>>>(Qb, Kb, bq /*unused*/, Sf, nullptr, DK, DK, DK);

  // 4) P = softmax(S, axis=0)                  (bf16 out)
  softmax_part<<<dim3(DK / 256, NC), blk, 0, stream>>>(Sf, Mp, Dp, DK, CH);
  softmax_write<<<dim3(DK / 256, NC), blk, 0, stream>>>(Sf, Mp, Dp, Pb, DK, CH, NC);

  // 5) logits = P @ Wv^T + bv                  (f32 out -> d_out)
  const dim3 gO(OUT_F / BN, DK / BM);
  gemm_bt<0, 0><<<gO, blk, 0, stream>>>(Pb, Wvb, bv, (float*)d_out, nullptr, DK, OUT_F, DK);
}
